// Model_46952582480570
// MI455X (gfx1250) — matrix operand order repaired and checked by reading, not compiled, not run on hardware
//
#include <hip/hip_runtime.h>
#include <stdint.h>

// LIF 2-layer SNN, T=50 steps, B=131072, HID=100.
// Wave-per-16-batch-tile, transposed (hidden on M, batch on N):
//   cur1^T = W1 @ x_t^T   -> 7x V_WMMA_F32_16X16X4_F32   (K=2 padded to 4)
//   cur2^T = W2 @ spk1^T  -> 4x V_WMMA_F32_16X16X32_F16  (K=112 padded to 128)
// mem1 state lives in the WMMA C-fragment layout in VGPRs for all 50 steps.
// Spikes (exact 0/1 in f16) persist packed across steps: they feed both the
// next step's reset term (via v_fma_mix_f32, f16 operand) and the layer-2
// WMMA B fragment (via ds_swizzle SWAPX16 half-exchange).

#define TSTEPS 50
#define BATCH  131072
#define HID    100
#define BETA   0.9f
#define THR    0.5f

typedef __attribute__((ext_vector_type(2)))  float    v2f;
typedef __attribute__((ext_vector_type(8)))  float    v8f;
typedef __attribute__((ext_vector_type(16))) _Float16 v16h;
typedef __attribute__((ext_vector_type(8)))  uint32_t v8u;

#define SWAPX16 0x401f   // ds_swizzle: xor=0x10, or=0, and=0x1f

// cm = cur - 0.5 * spk, spk taken as f16 from low/high half of packed dword.
__device__ __forceinline__ float fmix_sub_lo(uint32_t pk, float cur) {
  float r;
  asm("v_fma_mix_f32 %0, %1, -0.5, %2 op_sel:[0,0,0] op_sel_hi:[1,0,0]"
      : "=v"(r) : "v"(pk), "v"(cur));
  return r;
}
__device__ __forceinline__ float fmix_sub_hi(uint32_t pk, float cur) {
  float r;
  asm("v_fma_mix_f32 %0, %1, -0.5, %2 op_sel:[1,0,0] op_sel_hi:[1,0,0]"
      : "=v"(r) : "v"(pk), "v"(cur));
  return r;
}

__global__ void __launch_bounds__(256) snn_lif_kernel(
    const float* __restrict__ x,    // [T, BATCH, 2]
    const float* __restrict__ W1,   // [HID, 2]
    const float* __restrict__ W2,   // [2, HID]
    float* __restrict__ out)        // spk_rec [T,B,2] ++ mem_rec [T,B,2]
{
  const int lane = threadIdx.x & 31;
  const int wid  = (blockIdx.x * blockDim.x + threadIdx.x) >> 5; // global wave
  const int b0   = wid * 16;                                      // batch tile
  if (b0 >= BATCH) return;                                        // wave-uniform
  const bool lo  = lane < 16;
  const int  n   = lane & 15;

  // ---- W1 A-fragments: f32 A 16x4 per hidden tile (K=0,1 real; K=2,3 zero).
  // A layout: vgpr0 = {lanes0-15: K=0, lanes16-31: K=2}, vgpr1 = {K=1, K=3}.
  v2f w1a[7];
#pragma unroll
  for (int i = 0; i < 7; ++i) {
    const int h = i * 16 + n;
    float a0 = 0.f, a1 = 0.f;
    if (lo && h < HID) { a0 = W1[h * 2 + 0]; a1 = W1[h * 2 + 1]; }
    w1a[i][0] = a0; w1a[i][1] = a1;
  }

  // ---- W2 A-fragments: f16 A 16x32 per K-block (M=0,1 real rows, rest 0).
  v16h w2a[4];
#pragma unroll
  for (int kb = 0; kb < 4; ++kb) {
    v8u u = {};
#pragma unroll
    for (int v = 0; v < 8; ++v) {
      const int kloc = 2 * v + (lo ? 0 : 16);
      const int k = kb * 32 + kloc;
      float f0 = 0.f, f1 = 0.f;
      if (n < 2) {
        if (k     < HID) f0 = W2[n * HID + k];
        if (k + 1 < HID) f1 = W2[n * HID + k + 1];
      }
      const uint32_t h0 = (uint32_t)__builtin_bit_cast(uint16_t, (_Float16)f0);
      const uint32_t h1 = (uint32_t)__builtin_bit_cast(uint16_t, (_Float16)f1);
      u[v] = h0 | (h1 << 16);
    }
    w2a[kb] = __builtin_bit_cast(v16h, u);
  }

  // ---- Persistent state.
  v8f mem1[7] = {};
  float m2_0 = 0.f, m2_1 = 0.f;

  // Packed f16 spikes (0/1) per tile; persists across steps (= spk_prev).
  // pk[i][j]: low lanes hold (m=2j,2j+1), high lanes (m=8+2j,9+2j). Tile 7
  // (hidden 112..127 padding) is constant 0.
  uint32_t pk[8][4];
#pragma unroll
  for (int i = 0; i < 8; ++i)
#pragma unroll
    for (int j = 0; j < 4; ++j) pk[i][j] = 0u;

  const v8f zero8 = {};

  // Per-lane streaming pointers: all 32 lanes load x (hi lanes duplicate);
  // stores are full-wave: lanes 0-15 -> spk_rec, lanes 16-31 -> mem_rec.
  const float* xp = x + (size_t)(b0 + n) * 2;
  float* op = (lo ? out : out + (size_t)TSTEPS * BATCH * 2)
              + (size_t)(b0 + n) * 2;

  v2f xv = *(const v2f*)xp;  // software-pipelined x load for step 0

  for (int t = 0; t < TSTEPS; ++t) {
    // Issue next step's load early; its wait lands a full step of compute away.
    xp += (size_t)BATCH * 2;
    v2f xnext = {};
    if (t + 1 < TSTEPS) {
      xnext = *(const v2f*)xp;
      __builtin_prefetch(xp + (size_t)BATCH * 2, 0, 0);  // t+2, speculative
    }

    // x B-fragment (f32 B 4x16): vgpr0 = {K=0 | K=2}, vgpr1 = {K=1 | K=3};
    // K=2,3 padded with zero (high lanes).
    v2f xb;
    xb[0] = lo ? xv[0] : 0.f;
    xb[1] = lo ? xv[1] : 0.f;

    // Layer 1: WMMA cur1; reset folded via f16 spk_prev (v_fma_mix_f32);
    // LIF update in C-layout; new spikes packed to f16 in place.
#pragma unroll
    for (int i = 0; i < 7; ++i) {
      v8f cur = __builtin_amdgcn_wmma_f32_16x16x4_f32(
          false, w1a[i], false, xb, (short)0, zero8, false, false);
#pragma unroll
      for (int j = 0; j < 4; ++j) {
        const uint32_t pj = pk[i][j];
        const float cm0 = fmix_sub_lo(pj, cur[2 * j]);
        const float cm1 = fmix_sub_hi(pj, cur[2 * j + 1]);
        const float m0 = fmaf(BETA, mem1[i][2 * j],     cm0);
        const float m1 = fmaf(BETA, mem1[i][2 * j + 1], cm1);
        mem1[i][2 * j]     = m0;
        mem1[i][2 * j + 1] = m1;
        pk[i][j] = (m0 > THR ? 0x00003C00u : 0u)   // f16 1.0
                 | (m1 > THR ? 0x3C000000u : 0u);
      }
    }

    // Layer 2: build f16 B 32x16 fragments from C-layout spikes.
    // B layout: lanes0-15 K=2v,2v+1 ; lanes16-31 K=16+2v,16+2v+1 (N=lane%16).
    v8f c2 = {};
#pragma unroll
    for (int kb = 0; kb < 4; ++kb) {
      v8u bu = {};
#pragma unroll
      for (int v = 0; v < 8; ++v) {
        const int j = v & 3;
        const uint32_t ea = pk[2 * kb][j];
        const uint32_t eb = pk[2 * kb + 1][j];
        if (v < 4) {  // low: local even-tile; high: odd-tile from lane-16
          const uint32_t sw = (uint32_t)__builtin_amdgcn_ds_swizzle((int)eb, SWAPX16);
          bu[v] = lo ? ea : sw;
        } else {      // low: even-tile from lane+16; high: local odd-tile
          const uint32_t sw = (uint32_t)__builtin_amdgcn_ds_swizzle((int)ea, SWAPX16);
          bu[v] = lo ? sw : eb;
        }
      }
      c2 = __builtin_amdgcn_wmma_f32_16x16x32_f16(
          false, w2a[kb], false, __builtin_bit_cast(v16h, bu),
          (short)0, c2, false, false);
    }

    // Layer-2 LIF update: cur2^T rows M=0,1 in c2[0], c2[1] on lanes 0-15.
    {
      const float c20 = c2[0], c21 = c2[1];
      const float cm0 = (m2_0 > THR) ? (c20 - THR) : c20;
      const float cm1 = (m2_1 > THR) ? (c21 - THR) : c21;
      m2_0 = fmaf(BETA, m2_0, cm0);
      m2_1 = fmaf(BETA, m2_1, cm1);
      const float s20 = (m2_0 > THR) ? 1.f : 0.f;
      const float s21 = (m2_1 > THR) ? 1.f : 0.f;

      // Full-wave NT store: hi lanes pick up (m2_0,m2_1) from lane-16.
      const uint32_t sm0 = (uint32_t)__builtin_amdgcn_ds_swizzle(
          (int)__builtin_bit_cast(uint32_t, m2_0), SWAPX16);
      const uint32_t sm1 = (uint32_t)__builtin_amdgcn_ds_swizzle(
          (int)__builtin_bit_cast(uint32_t, m2_1), SWAPX16);
      v2f dv;
      dv[0] = lo ? s20 : __builtin_bit_cast(float, sm0);
      dv[1] = lo ? s21 : __builtin_bit_cast(float, sm1);
      __builtin_nontemporal_store(dv, (v2f*)op);
      op += (size_t)BATCH * 2;
    }

    xv = xnext;
  }
}

extern "C" void kernel_launch(void* const* d_in, const int* in_sizes, int n_in,
                              void* d_out, int out_size, void* d_ws, size_t ws_size,
                              hipStream_t stream) {
  (void)in_sizes; (void)n_in; (void)out_size; (void)d_ws; (void)ws_size;
  const float* x  = (const float*)d_in[0];
  const float* W1 = (const float*)d_in[1];
  const float* W2 = (const float*)d_in[2];
  float* out = (float*)d_out;

  // 16 batch per wave, 8 waves (256 threads) per block -> 128 batch per block.
  const int tiles  = BATCH / 16;            // 8192 waves
  const int block  = 256;
  const int grid   = tiles / (block / 32);  // 1024 blocks
  snn_lif_kernel<<<grid, block, 0, stream>>>(x, W1, W2, out);
}
